// MultiheadAttention_1202590843096
// MI455X (gfx1250) — matrix operand order repaired and checked by reading, not compiled, not run on hardware
//
#include <hip/hip_runtime.h>
#include <stdint.h>

#define BB 2
#define SS 2048
#define DD 512
#define HH 8
#define DKK 64

typedef __attribute__((ext_vector_type(16))) __bf16 bf16x16;
typedef __attribute__((ext_vector_type(8)))  __bf16 bf16x8;
typedef __attribute__((ext_vector_type(8)))  float  f32x8;
typedef __attribute__((ext_vector_type(4)))  int    i32x4;

typedef __attribute__((address_space(1))) i32x4 as1_i32x4;
typedef __attribute__((address_space(3))) i32x4 as3_i32x4;

#if defined(__has_builtin)
#if __has_builtin(__builtin_amdgcn_global_load_async_to_lds_b128) && \
    __has_builtin(__builtin_amdgcn_s_wait_asynccnt)
#define HAVE_ASYNC_LDS 1
#endif
#endif

// Stage one 16-byte chunk global -> LDS (async DMA if available).
static __device__ __forceinline__ void stage_b128(const __bf16* g, __bf16* l)
{
#ifdef HAVE_ASYNC_LDS
  __builtin_amdgcn_global_load_async_to_lds_b128(
      (as1_i32x4*)g, (as3_i32x4*)l, 0, 0);
#else
  i32x4 v = *(const i32x4*)g;
  *(i32x4*)l = v;
#endif
}

static __device__ __forceinline__ void stage_wait()
{
#ifdef HAVE_ASYNC_LDS
  __builtin_amdgcn_s_wait_asynccnt(0);
#endif
}

// ---------------------------------------------------------------------------
// Stage 0: one-shot fp32 -> bf16 conversion (activations + weights), so the
// GEMM inner loops are pure bf16 loads + WMMA with no per-K-step VALU.
// ---------------------------------------------------------------------------
__global__ void __launch_bounds__(256)
cvt_kernel(const float* __restrict__ in, __bf16* __restrict__ out, int n)
{
  int i = (blockIdx.x * blockDim.x + threadIdx.x) * 8;
  if (i < n) {
    f32x8 v = *(const f32x8*)(in + i);
    bf16x8 o;
#pragma unroll
    for (int j = 0; j < 8; ++j) o[j] = (__bf16)v[j];
    *(bf16x8*)(out + i) = o;
  }
}

// ---------------------------------------------------------------------------
// Stage 1: QKV projection.  out = X @ W^T + bias (X, W already bf16).
// One wave owns a 32x16 output tile (two accumulators share each B tile).
// mode 0/1: bf16 head-split [B,H,S,DK].  mode 2: v transposed [B,H,DK,S].
// ---------------------------------------------------------------------------
__global__ void __launch_bounds__(128)
proj_kernel(const __bf16* __restrict__ X, const __bf16* __restrict__ W,
            const float* __restrict__ bias, __bf16* __restrict__ out, int mode)
{
  const int lane = threadIdx.x & 31;
  const int wid  = threadIdx.x >> 5;
  const int m0 = blockIdx.x * 32;
  const int n0 = (blockIdx.y * 4 + wid) * 16;
  const int lo = lane & 15;
  const int hi = lane >> 4;

  f32x8 acc0 = {}, acc1 = {};
  const __bf16* xrow0 = X + (size_t)(m0 + lo) * DD;
  const __bf16* xrow1 = X + (size_t)(m0 + 16 + lo) * DD;
  const __bf16* wrow  = W + (size_t)(n0 + lo) * DD;

  for (int k0 = 0; k0 < DD; k0 += 32) {
    bf16x16 bm = *(const bf16x16*)(wrow + k0 + hi * 16);
    bf16x8 a0l = *(const bf16x8*)(xrow0 + k0 + hi * 16);
    bf16x8 a0h = *(const bf16x8*)(xrow0 + k0 + 8 + hi * 16);
    bf16x8 a1l = *(const bf16x8*)(xrow1 + k0 + hi * 16);
    bf16x8 a1h = *(const bf16x8*)(xrow1 + k0 + 8 + hi * 16);
    bf16x16 a0 = __builtin_shufflevector(a0l, a0h, 0,1,2,3,4,5,6,7,8,9,10,11,12,13,14,15);
    bf16x16 a1 = __builtin_shufflevector(a1l, a1h, 0,1,2,3,4,5,6,7,8,9,10,11,12,13,14,15);
    acc0 = __builtin_amdgcn_wmma_f32_16x16x32_bf16(false, a0, false, bm,
                                                   (short)0, acc0, false, false);
    acc1 = __builtin_amdgcn_wmma_f32_16x16x32_bf16(false, a1, false, bm,
                                                   (short)0, acc1, false, false);
  }

  const float bv = bias[n0 + lo];
  const int n  = n0 + lo;
  const int hh = n >> 6;
  const int dk = n & 63;
#pragma unroll
  for (int mt = 0; mt < 2; ++mt) {
#pragma unroll
    for (int r = 0; r < 8; ++r) {
      int rIdx = m0 + mt * 16 + r + 8 * hi;   // C/D: M = r + 8*(lane>=16)
      int b = rIdx >> 11;
      int s = rIdx & (SS - 1);
      float v = (mt ? acc1[r] : acc0[r]) + bv;
      __bf16 val = (__bf16)v;
      if (mode == 2)
        out[((size_t)((b * HH + hh) * DKK + dk)) * SS + s] = val;   // v^T
      else
        out[((size_t)((b * HH + hh) * SS + s)) * DKK + dk] = val;   // q, k
    }
  }
}

// ---------------------------------------------------------------------------
// Stage 2: flash attention with multiplicative sph mask on scores.
// Block = 8 waves sharing one (b,h); K / V^T tiles are staged into LDS once
// per block (async global->LDS when available, double-buffered) and consumed
// by all waves via ds_load_b128.  P converts C->A layout through LDS.
// ---------------------------------------------------------------------------
#define WPB 8

__global__ void __launch_bounds__(256)
attn_kernel(const __bf16* __restrict__ Q, const __bf16* __restrict__ Km,
            const __bf16* __restrict__ VT, const float* __restrict__ SPH,
            __bf16* __restrict__ O)
{
  __shared__ __align__(32) __bf16 kbuf[2][32 * 64];   // K tile: 32 kv x 64 dk
  __shared__ __align__(32) __bf16 vbuf[2][64 * 32];   // V^T tile: 64 dk x 32 kv
  __shared__ __align__(32) __bf16 ptile[WPB][16 * 32];

  const int tid  = threadIdx.x;
  const int lane = tid & 31;
  const int wid  = tid >> 5;
  const int lo = lane & 15;
  const int hi = lane >> 4;

  const int bh = blockIdx.y;       // b*H + h
  const int b  = bh >> 3;
  const int h  = bh & 7;
  const int q0 = (blockIdx.x * WPB + wid) * 16;

  const __bf16* qbase = Q  + (size_t)bh * SS * DKK;
  const __bf16* kbase = Km + (size_t)bh * SS * DKK;
  const __bf16* vtb   = VT + (size_t)bh * DKK * SS;
  const float*  sphb  = SPH + (size_t)b * SS * SS;

  // cooperative staging offsets: 256 threads x 16B cover each 4KB tile
  const int krow = tid >> 3, kcol = (tid & 7) * 8;    // K: 32 rows x 64 halfs
  const int vrow = tid >> 2, vcol = (tid & 3) * 8;    // V: 64 rows x 32 halfs

  // q A-tiles for K = 0..31 and 32..63 (DK = 64)
  bf16x16 a0, a1;
  {
    const __bf16* qr = qbase + (size_t)(q0 + lo) * DKK;
    bf16x8 c0 = *(const bf16x8*)(qr + hi * 16);
    bf16x8 c1 = *(const bf16x8*)(qr + 8 + hi * 16);
    bf16x8 c2 = *(const bf16x8*)(qr + 32 + hi * 16);
    bf16x8 c3 = *(const bf16x8*)(qr + 40 + hi * 16);
    a0 = __builtin_shufflevector(c0, c1, 0,1,2,3,4,5,6,7,8,9,10,11,12,13,14,15);
    a1 = __builtin_shufflevector(c2, c3, 0,1,2,3,4,5,6,7,8,9,10,11,12,13,14,15);
  }

  float mrun[8], lrun[8];
  f32x8 accd[4] = {f32x8{}, f32x8{}, f32x8{}, f32x8{}};
#pragma unroll
  for (int r = 0; r < 8; ++r) { mrun[r] = -__builtin_inff(); lrun[r] = 0.f; }

  __bf16* myl = &ptile[wid][0];

  // prologue: stage tile j0 = 0 into buffer 0
  stage_b128(kbase + (size_t)krow * DKK + kcol, &kbuf[0][krow * 64 + kcol]);
  stage_b128(vtb + (size_t)vrow * SS + vcol,    &vbuf[0][vrow * 32 + vcol]);
  stage_wait();
  __syncthreads();

  for (int j0 = 0; j0 < SS; j0 += 32) {
    const int cur = (j0 >> 5) & 1;
    const int nxt = cur ^ 1;

    // issue async staging of next tile (overlaps with compute below)
    if (j0 + 32 < SS) {
      stage_b128(kbase + (size_t)(j0 + 32 + krow) * DKK + kcol,
                 &kbuf[nxt][krow * 64 + kcol]);
      stage_b128(vtb + (size_t)vrow * SS + (j0 + 32) + vcol,
                 &vbuf[nxt][vrow * 32 + vcol]);
      __builtin_prefetch((const void*)(sphb + (size_t)(q0 + lo) * SS + j0 + 32), 0, 1);
    }

    // scores tile 16x32 from LDS K tile: two 16x16 C accumulators
    f32x8 c[2];
#pragma unroll
    for (int ct = 0; ct < 2; ++ct) {
      const __bf16* kr = &kbuf[cur][(16 * ct + lo) * 64];
      bf16x16 b0 = *(const bf16x16*)(kr + hi * 16);
      bf16x16 b1 = *(const bf16x16*)(kr + 32 + hi * 16);
      f32x8 t = {};
      t = __builtin_amdgcn_wmma_f32_16x16x32_bf16(false, a0, false, b0,
                                                  (short)0, t, false, false);
      t = __builtin_amdgcn_wmma_f32_16x16x32_bf16(false, a1, false, b1,
                                                  (short)0, t, false, false);
      c[ct] = t;
    }

    // scale + multiplicative sph mask (non-temporal: streamed exactly once)
    float s0[8], s1[8];
#pragma unroll
    for (int r = 0; r < 8; ++r) {
      int m = r + 8 * hi;
      const float* sp = sphb + (size_t)(q0 + m) * SS + j0 + lo;
      s0[r] = c[0][r] * 0.125f * __builtin_nontemporal_load(sp);
      s1[r] = c[1][r] * 0.125f * __builtin_nontemporal_load(sp + 16);
    }

    // online softmax; row stats via 16-lane xor-shuffle reductions
#pragma unroll
    for (int r = 0; r < 8; ++r) {
      float tm = fmaxf(s0[r], s1[r]);
#pragma unroll
      for (int msk = 1; msk < 16; msk <<= 1)
        tm = fmaxf(tm, __shfl_xor(tm, msk, 32));
      float mn   = fmaxf(mrun[r], tm);
      float corr = __expf(mrun[r] - mn);
      float p0   = __expf(s0[r] - mn);
      float p1   = __expf(s1[r] - mn);
      float rs = p0 + p1;
#pragma unroll
      for (int msk = 1; msk < 16; msk <<= 1)
        rs += __shfl_xor(rs, msk, 32);
      lrun[r] = lrun[r] * corr + rs;
      mrun[r] = mn;
#pragma unroll
      for (int t = 0; t < 4; ++t) accd[t][r] *= corr;
      int m = r + 8 * hi;
      myl[m * 32 + lo]      = (__bf16)p0;   // C-layout -> LDS row-major
      myl[m * 32 + 16 + lo] = (__bf16)p1;
    }

    // reload P in A-layout from LDS (same-wave DS ops are in-order)
    bf16x16 pa;
    {
      const __bf16* pr = myl + lo * 32;
      bf16x8 pc0 = *(const bf16x8*)(pr + hi * 16);
      bf16x8 pc1 = *(const bf16x8*)(pr + 8 + hi * 16);
      pa = __builtin_shufflevector(pc0, pc1, 0,1,2,3,4,5,6,7,8,9,10,11,12,13,14,15);
    }

    // out += P @ V : four 16-wide d tiles, B from LDS V^T tile
#pragma unroll
    for (int t = 0; t < 4; ++t) {
      const __bf16* vr = &vbuf[cur][(t * 16 + lo) * 32 + hi * 16];
      bf16x16 vb = *(const bf16x16*)vr;
      accd[t] = __builtin_amdgcn_wmma_f32_16x16x32_bf16(false, pa, false, vb,
                                                        (short)0, accd[t],
                                                        false, false);
    }

    // next tile's async copies must land; everyone done reading cur
    stage_wait();
    __syncthreads();
  }

  // normalize, store bf16 [B,S,D] for the output projection
#pragma unroll
  for (int r = 0; r < 8; ++r) {
    int m = r + 8 * hi;
    float inv = 1.0f / lrun[r];
    size_t base = ((size_t)(b * SS + q0 + m)) * DD + h * DKK;
#pragma unroll
    for (int t = 0; t < 4; ++t)
      O[base + t * 16 + lo] = (__bf16)(accd[t][r] * inv);
  }
}

// ---------------------------------------------------------------------------
// Stage 3: output projection.  out = O_bf16 @ Wo^T + bo, fp32 store.
// ---------------------------------------------------------------------------
__global__ void __launch_bounds__(128)
outproj_kernel(const __bf16* __restrict__ Xb, const __bf16* __restrict__ W,
               const float* __restrict__ bias, float* __restrict__ out)
{
  const int lane = threadIdx.x & 31;
  const int wid  = threadIdx.x >> 5;
  const int m0 = blockIdx.x * 32;
  const int n0 = (blockIdx.y * 4 + wid) * 16;
  const int lo = lane & 15;
  const int hi = lane >> 4;

  f32x8 acc0 = {}, acc1 = {};
  const __bf16* xrow0 = Xb + (size_t)(m0 + lo) * DD;
  const __bf16* xrow1 = Xb + (size_t)(m0 + 16 + lo) * DD;
  const __bf16* wrow  = W  + (size_t)(n0 + lo) * DD;

  for (int k0 = 0; k0 < DD; k0 += 32) {
    bf16x16 bm = *(const bf16x16*)(wrow + k0 + hi * 16);
    bf16x8 a0l = *(const bf16x8*)(xrow0 + k0 + hi * 16);
    bf16x8 a0h = *(const bf16x8*)(xrow0 + k0 + 8 + hi * 16);
    bf16x8 a1l = *(const bf16x8*)(xrow1 + k0 + hi * 16);
    bf16x8 a1h = *(const bf16x8*)(xrow1 + k0 + 8 + hi * 16);
    bf16x16 a0 = __builtin_shufflevector(a0l, a0h, 0,1,2,3,4,5,6,7,8,9,10,11,12,13,14,15);
    bf16x16 a1 = __builtin_shufflevector(a1l, a1h, 0,1,2,3,4,5,6,7,8,9,10,11,12,13,14,15);
    acc0 = __builtin_amdgcn_wmma_f32_16x16x32_bf16(false, a0, false, bm,
                                                   (short)0, acc0, false, false);
    acc1 = __builtin_amdgcn_wmma_f32_16x16x32_bf16(false, a1, false, bm,
                                                   (short)0, acc1, false, false);
  }
  const float bv = bias[n0 + lo];
#pragma unroll
  for (int mt = 0; mt < 2; ++mt) {
#pragma unroll
    for (int r = 0; r < 8; ++r) {
      int rIdx = m0 + mt * 16 + r + 8 * hi;
      out[(size_t)rIdx * DD + n0 + lo] = (mt ? acc1[r] : acc0[r]) + bv;
    }
  }
}

// ---------------------------------------------------------------------------
extern "C" void kernel_launch(void* const* d_in, const int* in_sizes, int n_in,
                              void* d_out, int out_size, void* d_ws, size_t ws_size,
                              hipStream_t stream)
{
  const float* query = (const float*)d_in[0];
  const float* key   = (const float*)d_in[1];
  const float* value = (const float*)d_in[2];
  const float* sph   = (const float*)d_in[3];
  const float* Wq    = (const float*)d_in[4];
  const float* bq    = (const float*)d_in[5];
  const float* Wk    = (const float*)d_in[6];
  const float* bk    = (const float*)d_in[7];
  const float* Wv    = (const float*)d_in[8];
  const float* bv    = (const float*)d_in[9];
  const float* Wo    = (const float*)d_in[10];
  const float* bo    = (const float*)d_in[11];
  float* out = (float*)d_out;

  const size_t elems = (size_t)BB * SS * DD;   // 2M elements per activation
  const size_t welms = (size_t)DD * DD;        // 256K elements per weight
  __bf16* p = (__bf16*)d_ws;
  __bf16* xq = p; p += elems;
  __bf16* xk = p; p += elems;
  __bf16* xv = p; p += elems;
  __bf16* wq = p; p += welms;
  __bf16* wk = p; p += welms;
  __bf16* wv = p; p += welms;
  __bf16* wo = p; p += welms;
  __bf16* qs = p; p += elems;                  // q   [B,H,S,DK]
  __bf16* ks = p; p += elems;                  // k   [B,H,S,DK]
  __bf16* vt = p; p += elems;                  // v^T [B,H,DK,S]
  __bf16* os = p; p += elems;                  // attn out [B,S,D]

  dim3 cb(256);
  dim3 cga((int)(elems / 8 / 256));
  dim3 cgw((int)(welms / 8 / 256));
  cvt_kernel<<<cga, cb, 0, stream>>>(query, xq, (int)elems);
  cvt_kernel<<<cga, cb, 0, stream>>>(key,   xk, (int)elems);
  cvt_kernel<<<cga, cb, 0, stream>>>(value, xv, (int)elems);
  cvt_kernel<<<cgw, cb, 0, stream>>>(Wq, wq, (int)welms);
  cvt_kernel<<<cgw, cb, 0, stream>>>(Wk, wk, (int)welms);
  cvt_kernel<<<cgw, cb, 0, stream>>>(Wv, wv, (int)welms);
  cvt_kernel<<<cgw, cb, 0, stream>>>(Wo, wo, (int)welms);

  dim3 pblock(128);
  dim3 pgrid(BB * SS / 32, DD / 64);           // (128, 8)
  proj_kernel<<<pgrid, pblock, 0, stream>>>(xq, wq, bq, qs, 0);
  proj_kernel<<<pgrid, pblock, 0, stream>>>(xk, wk, bk, ks, 1);
  proj_kernel<<<pgrid, pblock, 0, stream>>>(xv, wv, bv, vt, 2);

  dim3 ablock(256);
  dim3 agrid(SS / 16 / WPB, BB * HH);          // (16, 16)
  attn_kernel<<<agrid, ablock, 0, stream>>>(qs, ks, vt, sph, os);

  outproj_kernel<<<pgrid, pblock, 0, stream>>>(os, wo, bo, out);
}
